// MultiHeadAttention_54400055771154
// MI455X (gfx1250) — matrix operand order repaired and checked by reading, not compiled, not run on hardware
//
#include <hip/hip_runtime.h>

// MI455X / gfx1250: wave32, WMMA bf16 16x16x32, double-buffered TDM staging.
// MHA forward: f32->bf16 convert (once per tensor) -> QKV GEMMs (bf16 out) ->
// causal flash attention (bf16 out) -> output projection (f32 out).
// B=2, L=2048, D=1024, H=16, DK=64.

typedef __bf16 bf16;
typedef __attribute__((ext_vector_type(16))) __bf16 v16bf;
typedef __attribute__((ext_vector_type(8)))  __bf16 v8bf;
typedef __attribute__((ext_vector_type(8)))  float  v8f;
typedef unsigned int u32x4 __attribute__((ext_vector_type(4)));
typedef int          i32x8 __attribute__((ext_vector_type(8)));
typedef int          i32x4 __attribute__((ext_vector_type(4)));

#define B_  2
#define L_  2048
#define D_  1024
#define H_  16
#define DK_ 64
#define M_  (B_ * L_)   // 4096

#if __has_builtin(__builtin_amdgcn_tensor_load_to_lds) && __has_builtin(__builtin_amdgcn_s_wait_tensorcnt)
#define HAS_TDM 1
#else
#define HAS_TDM 0
#endif

__device__ __forceinline__ bf16 f2bf(float x) { return (bf16)x; }

__device__ __forceinline__ v8f wmma_bf16(v16bf a, v16bf b, v8f c) {
    return __builtin_amdgcn_wmma_f32_16x16x32_bf16(
        /*neg_a=*/false, a, /*neg_b=*/false, b,
        /*c_mod=*/(short)0, c, /*reuse_a=*/false, /*reuse_b=*/false);
}

// A-fragment (16x32 bf16, MxK), ISA 7.12.2: lanes 0-15 row M=lane, K={0..7,16..23};
// lanes 16-31 row M=lane-16, K={8..15,24..31}. tile row-major, leading dim ld.
__device__ __forceinline__ v16bf load_a_frag(const bf16* tile, int ld, int row0, int col0) {
    const int lane = threadIdx.x & 31;
    const int half = lane >> 4;
    const int r    = lane & 15;
    const bf16* p = tile + (size_t)(row0 + r) * ld + col0 + half * 8;
    v8bf lo = *(const v8bf*)(p);
    v8bf hi = *(const v8bf*)(p + 16);
    v16bf f;
#pragma unroll
    for (int i = 0; i < 8; ++i) { f[i] = lo[i]; f[i + 8] = hi[i]; }
    return f;
}

// B-fragment (32x16 bf16, KxN): lane n holds column n, K = {half*8..+7, 16+half*8..+7}
// (matching the A-fragment K order). tile stores B^T row-major: tile[n][k].
__device__ __forceinline__ v16bf load_b_frag(const bf16* tile, int ld, int row0, int col0) {
    const int lane = threadIdx.x & 31;
    const int half = lane >> 4;
    const int n    = lane & 15;
    const bf16* p = tile + (size_t)(row0 + n) * ld + col0 + half * 8;
    v8bf lo = *(const v8bf*)(p);
    v8bf hi = *(const v8bf*)(p + 16);
    v16bf f;
#pragma unroll
    for (int i = 0; i < 8; ++i) { f[i] = lo[i]; f[i + 8] = hi[i]; }
    return f;
}

// ---------------------------------------------------------------------------
// TDM: issue a 2D bf16 tile load (tile_w x tile_h) from row-major global into
// LDS at lds_byte_off, with LDS row padding (stride 72 elems for 64-elem rows:
// 32-DWORD rows -> pad_interval enc 4, 4-DWORD pad -> pad_amount enc 3).
// D# layout per CDNA5 ISA (2D: groups 2/3 zero).
// ---------------------------------------------------------------------------
__device__ __forceinline__ void tdm_load_tile_2d_bf16(
    const bf16* gaddr, unsigned lds_byte_off,
    unsigned tensor_w, unsigned tensor_h, unsigned row_stride_elems,
    unsigned tile_w, unsigned tile_h) {
#if HAS_TDM
    const unsigned long long ga = (unsigned long long)(uintptr_t)gaddr;
    u32x4 g0;
    g0[0] = 1u;                                             // count=1, user D#
    g0[1] = lds_byte_off;                                   // lds_addr
    g0[2] = (unsigned)(ga & 0xFFFFFFFFu);                   // global_addr[31:0]
    g0[3] = (unsigned)((ga >> 32) & 0x01FFFFFFu) | (2u << 30);  // addr[56:32], type=2
    i32x8 g1;
    g1[0] = (int)((1u << 16) | (1u << 20) | (4u << 22) | (3u << 25));
            // data_size=2B, pad_enable, pad_interval=32DW, pad_amount=4DW
    g1[1] = (int)((tensor_w & 0xFFFFu) << 16);              // tensor_dim0[15:0]
    g1[2] = (int)(((tensor_w >> 16) & 0xFFFFu) | ((tensor_h & 0xFFFFu) << 16));
    g1[3] = (int)(((tensor_h >> 16) & 0xFFFFu) | ((tile_w & 0xFFFFu) << 16));
    g1[4] = (int)(tile_h & 0xFFFFu);                        // tile_dim1; tile_dim2=0
    g1[5] = (int)row_stride_elems;                          // tensor_dim0_stride[31:0]
    g1[6] = 0;
    g1[7] = 0;
    i32x4 z4; z4[0] = 0; z4[1] = 0; z4[2] = 0; z4[3] = 0;
    i32x8 z8;
#pragma unroll
    for (int i = 0; i < 8; ++i) z8[i] = 0;
    __builtin_amdgcn_tensor_load_to_lds(g0, g1, z4, z4, z8, 0);
#else
    (void)gaddr; (void)lds_byte_off; (void)tensor_w; (void)tensor_h;
    (void)row_stride_elems; (void)tile_w; (void)tile_h;
#endif
}

// ---------------------------------------------------------------------------
// Elementwise f32 -> bf16 (8 elements/thread, b128 in / b128 out).
// ---------------------------------------------------------------------------
__global__ __launch_bounds__(256)
void cvt_f32_bf16_kernel(const float* __restrict__ in, bf16* __restrict__ out, int n) {
    const int i = (blockIdx.x * 256 + threadIdx.x) * 8;
    if (i >= n) return;
    const float4 a = *(const float4*)(in + i);
    const float4 b = *(const float4*)(in + i + 4);
    v8bf o;
    o[0] = f2bf(a.x); o[1] = f2bf(a.y); o[2] = f2bf(a.z); o[3] = f2bf(a.w);
    o[4] = f2bf(b.x); o[5] = f2bf(b.y); o[6] = f2bf(b.z); o[7] = f2bf(b.w);
    *(v8bf*)(out + i) = o;
}

// ---------------------------------------------------------------------------
// GEMM: Y = X @ W^T + bias. X:(M,K) bf16 row-major, W:(N,K) bf16 row-major.
// out_mode 0: Y f32 row-major (M,N). out_mode 1: Y bf16 in (B,H,L,DK).
// Block 128 threads (4 waves), tile 64x64, K-step 64.
// Double-buffered TDM: issue tile i+1, s_wait_tensorcnt 2 retires tile i
// (TDM completes in order), so DMA overlaps the 8 WMMAs of the current tile.
// ---------------------------------------------------------------------------
__global__ __launch_bounds__(128)
void mha_gemm_bias_kernel(const bf16* __restrict__ X, const bf16* __restrict__ W,
                          const float* __restrict__ bias, void* __restrict__ Yv,
                          int M, int N, int K, int out_mode) {
    __shared__ __align__(16) bf16 sA[2][64][72];
    __shared__ __align__(16) bf16 sB[2][64][72];

    const int tid  = threadIdx.x;
    const int wave = tid >> 5;
    const int lane = tid & 31;
    const int bm = blockIdx.y * 64;
    const int bn = blockIdx.x * 64;
    const int wm = (wave >> 1) * 32;
    const int wn = (wave & 1) * 32;

    v8f acc[2][2];
#pragma unroll
    for (int i = 0; i < 2; ++i)
#pragma unroll
        for (int j = 0; j < 2; ++j)
#pragma unroll
            for (int e = 0; e < 8; ++e) acc[i][j][e] = 0.0f;

#if HAS_TDM
    if (tid < 32) {   // prologue: fill buffer 0
        tdm_load_tile_2d_bf16(X + (size_t)bm * K,
                              (unsigned)(uintptr_t)&sA[0][0][0], K, M - bm, K, 64, 64);
        tdm_load_tile_2d_bf16(W + (size_t)bn * K,
                              (unsigned)(uintptr_t)&sB[0][0][0], K, N - bn, K, 64, 64);
    }
#endif

    for (int k0 = 0; k0 < K; k0 += 64) {
        const int buf = (k0 >> 6) & 1;
#if HAS_TDM
        if (tid < 32) {
            if (k0 + 64 < K) {   // issue next tile into the other buffer
                tdm_load_tile_2d_bf16(X + (size_t)bm * K + k0 + 64,
                                      (unsigned)(uintptr_t)&sA[buf ^ 1][0][0],
                                      K, M - bm, K, 64, 64);
                tdm_load_tile_2d_bf16(W + (size_t)bn * K + k0 + 64,
                                      (unsigned)(uintptr_t)&sB[buf ^ 1][0][0],
                                      K, N - bn, K, 64, 64);
                __builtin_amdgcn_s_wait_tensorcnt(2);   // current pair retired
            } else {
                __builtin_amdgcn_s_wait_tensorcnt(0);
            }
        }
#else
#pragma unroll
        for (int i = 0; i < 4; ++i) {
            int c = tid + 128 * i;           // 512 chunks of 8 bf16
            int row = c >> 3;
            int c8  = (c & 7) * 8;
            *(v8bf*)(&sA[buf][row][c8]) = *(const v8bf*)(X + (size_t)(bm + row) * K + k0 + c8);
            *(v8bf*)(&sB[buf][row][c8]) = *(const v8bf*)(W + (size_t)(bn + row) * K + k0 + c8);
        }
#endif
        __syncthreads();
#pragma unroll
        for (int kk = 0; kk < 64; kk += 32) {
            v16bf a0 = load_a_frag(&sA[buf][0][0], 72, wm + 0, kk);
            v16bf a1 = load_a_frag(&sA[buf][0][0], 72, wm + 16, kk);
            v16bf b0 = load_b_frag(&sB[buf][0][0], 72, wn + 0, kk);
            v16bf b1 = load_b_frag(&sB[buf][0][0], 72, wn + 16, kk);
            acc[0][0] = wmma_bf16(a0, b0, acc[0][0]);
            acc[0][1] = wmma_bf16(a0, b1, acc[0][1]);
            acc[1][0] = wmma_bf16(a1, b0, acc[1][0]);
            acc[1][1] = wmma_bf16(a1, b1, acc[1][1]);
        }
        __syncthreads();
    }

    // Epilogue. C/D layout: VGPR r -> M = r + 8*(lane/16), N = lane%16.
    const int half = lane >> 4;
    const int nl   = lane & 15;
#pragma unroll
    for (int mi = 0; mi < 2; ++mi) {
#pragma unroll
        for (int ni = 0; ni < 2; ++ni) {
            const int gn = bn + wn + ni * 16 + nl;
            const float bv = bias[gn];
#pragma unroll
            for (int r = 0; r < 8; ++r) {
                const int gm = bm + wm + mi * 16 + r + half * 8;
                const float val = acc[mi][ni][r] + bv;
                if (out_mode == 0) {
                    ((float*)Yv)[(size_t)gm * N + gn] = val;
                } else {
                    const int b = gm >> 11, l = gm & (L_ - 1);
                    const int h = gn >> 6, dk = gn & (DK_ - 1);
                    ((bf16*)Yv)[(((size_t)(b * H_ + h)) * L_ + l) * DK_ + dk] = f2bf(val);
                }
            }
        }
    }
}

// ---------------------------------------------------------------------------
// Flash attention, causal. Q/K/V: (B*H, L, DK) bf16. Output: (B,L,D) bf16.
// Block 128 threads = 4 waves, 64 queries/block. Grid: (L/64, B*H).
// K tiles double-buffered through the TDM; V staged transposed by the waves
// (TDM cannot transpose), overlapping the in-flight DMA.
// ---------------------------------------------------------------------------
__global__ __launch_bounds__(128)
void mha_flash_attn_kernel(const bf16* __restrict__ Q, const bf16* __restrict__ Km,
                           const bf16* __restrict__ V, bf16* __restrict__ O) {
    __shared__ __align__(16) bf16 sQ [64][72];
    __shared__ __align__(16) bf16 sK [2][64][72];
    __shared__ __align__(16) bf16 sVt[64][72];   // transposed: sVt[dk][key]
    __shared__ __align__(16) bf16 sP [64][72];   // wave-private P rows

    const int tid  = threadIdx.x;
    const int wave = tid >> 5;
    const int lane = tid & 31;
    const int half = lane >> 4;
    const int nl   = lane & 15;

    const int bh = blockIdx.y;
    const int b  = bh >> 4, h = bh & (H_ - 1);
    const int q0 = blockIdx.x * 64;
    const size_t base = (size_t)bh * L_ * DK_;
    const int nkb = (q0 >> 6) + 1;   // causal

    const float scale = 0.125f;   // 1/sqrt(DK), folded into score read
    const float NEG   = -3.0e38f;

#if HAS_TDM
    if (tid < 32) {   // prologue: Q tile + first K tile
        tdm_load_tile_2d_bf16(Q + base + (size_t)q0 * DK_,
                              (unsigned)(uintptr_t)&sQ[0][0], DK_, 64, DK_, 64, 64);
        tdm_load_tile_2d_bf16(Km + base,
                              (unsigned)(uintptr_t)&sK[0][0][0], DK_, 64, DK_, 64, 64);
    }
#else
#pragma unroll
    for (int i = 0; i < 4; ++i) {
        int c = tid + 128 * i; int row = c >> 3; int c8 = (c & 7) * 8;
        *(v8bf*)(&sQ[row][c8]) = *(const v8bf*)(Q + base + (size_t)(q0 + row) * DK_ + c8);
    }
#endif

    float m_i[8], l_i[8];
    v8f o_acc[4];
#pragma unroll
    for (int r = 0; r < 8; ++r) { m_i[r] = NEG; l_i[r] = 0.0f; }
#pragma unroll
    for (int dt = 0; dt < 4; ++dt)
#pragma unroll
        for (int e = 0; e < 8; ++e) o_acc[dt][e] = 0.0f;

    for (int kb = 0; kb < nkb; ++kb) {
        const int kstart = kb * 64;
        const int buf = kb & 1;
        __syncthreads();             // prior reads of sVt / sK[buf^1] done
#if HAS_TDM
        if (tid < 32) {
            if (kb + 1 < nkb) {      // prefetch next K tile into other buffer
                tdm_load_tile_2d_bf16(Km + base + (size_t)(kstart + 64) * DK_,
                                      (unsigned)(uintptr_t)&sK[buf ^ 1][0][0],
                                      DK_, 64, DK_, 64, 64);
            }
        }
#else
#pragma unroll
        for (int i = 0; i < 4; ++i) {
            int c = tid + 128 * i; int row = c >> 3; int c8 = (c & 7) * 8;
            *(v8bf*)(&sK[buf][row][c8]) =
                *(const v8bf*)(Km + base + (size_t)(kstart + row) * DK_ + c8);
        }
#endif
        // V staged transposed, overlapping the in-flight DMA.
#pragma unroll
        for (int i = 0; i < 4; ++i) {
            int c = tid + 128 * i; int row = c >> 3; int c8 = (c & 7) * 8;
            v8bf vv = *(const v8bf*)(V + base + (size_t)(kstart + row) * DK_ + c8);
#pragma unroll
            for (int e = 0; e < 8; ++e) sVt[c8 + e][row] = vv[e];
        }
#if HAS_TDM
        if (tid < 32) {
            if (kb + 1 < nkb) __builtin_amdgcn_s_wait_tensorcnt(1);  // current K done
            else              __builtin_amdgcn_s_wait_tensorcnt(0);
        }
#endif
        __syncthreads();

        // S = Q K^T for this wave's 16 rows (4 key sub-tiles, DK in 2 WMMA steps).
        v8f s[4];
#pragma unroll
        for (int t = 0; t < 4; ++t) {
#pragma unroll
            for (int e = 0; e < 8; ++e) s[t][e] = 0.0f;
#pragma unroll
            for (int kk = 0; kk < 64; kk += 32) {
                v16bf af = load_a_frag(&sQ[0][0], 72, wave * 16, kk);
                v16bf bf = load_b_frag(&sK[buf][0][0], 72, t * 16, kk);
                s[t] = wmma_bf16(af, bf, s[t]);
            }
        }

        // Causal mask + online softmax (row = one VGPR index across 16 lanes).
#pragma unroll
        for (int r = 0; r < 8; ++r) {
            const int qidx = q0 + wave * 16 + r + half * 8;
            float sv[4];
            float rowmax = NEG;
#pragma unroll
            for (int t = 0; t < 4; ++t) {
                const int kidx = kstart + t * 16 + nl;
                float x = s[t][r] * scale;
                x = (kidx > qidx) ? NEG : x;
                sv[t] = x;
                rowmax = fmaxf(rowmax, x);
            }
#pragma unroll
            for (int off = 1; off < 16; off <<= 1)
                rowmax = fmaxf(rowmax, __shfl_xor(rowmax, off, 32));
            const float mnew  = fmaxf(m_i[r], rowmax);
            const float alpha = __expf(m_i[r] - mnew);
            float psum = 0.0f;
#pragma unroll
            for (int t = 0; t < 4; ++t) {
                const float p = __expf(sv[t] - mnew);
                psum += p;
                sP[wave * 16 + r + half * 8][t * 16 + nl] = f2bf(p);
            }
#pragma unroll
            for (int off = 1; off < 16; off <<= 1)
                psum += __shfl_xor(psum, off, 32);
            l_i[r] = l_i[r] * alpha + psum;
            m_i[r] = mnew;
#pragma unroll
            for (int dt = 0; dt < 4; ++dt) o_acc[dt][r] *= alpha;
        }

        // O += P @ V (B operand from transposed V).
#pragma unroll
        for (int dt = 0; dt < 4; ++dt) {
#pragma unroll
            for (int kk = 0; kk < 64; kk += 32) {
                v16bf af = load_a_frag(&sP[0][0], 72, wave * 16, kk);
                v16bf bf = load_b_frag(&sVt[0][0], 72, dt * 16, kk);
                o_acc[dt] = wmma_bf16(af, bf, o_acc[dt]);
            }
        }
    }

    // Normalize, write bf16 to (B, L, D) with d = h*DK + dk.
#pragma unroll
    for (int r = 0; r < 8; ++r) {
        const int qidx = q0 + wave * 16 + r + half * 8;
        const float inv = 1.0f / l_i[r];
#pragma unroll
        for (int dt = 0; dt < 4; ++dt) {
            const int dk = dt * 16 + nl;
            O[((size_t)(b * L_ + qidx)) * D_ + h * DK_ + dk] = f2bf(o_acc[dt][r] * inv);
        }
    }
}

extern "C" void kernel_launch(void* const* d_in, const int* in_sizes, int n_in,
                              void* d_out, int out_size, void* d_ws, size_t ws_size,
                              hipStream_t stream) {
    const float* x  = (const float*)d_in[0];
    const float* Wq = (const float*)d_in[1];
    const float* bq = (const float*)d_in[2];
    const float* Wk = (const float*)d_in[3];
    const float* bk = (const float*)d_in[4];
    const float* Wv = (const float*)d_in[5];
    const float* bv = (const float*)d_in[6];
    const float* Wo = (const float*)d_in[7];
    const float* bo = (const float*)d_in[8];
    float* out = (float*)d_out;
    bf16* wsb  = (bf16*)d_ws;

    const size_t MD = (size_t)M_ * D_;   // 4M elements
    const size_t DD = (size_t)D_ * D_;   // 1M elements
    bf16* xb  = wsb;            // 4M
    bf16* wqb = xb  + MD;       // 1M
    bf16* wkb = wqb + DD;
    bf16* wvb = wkb + DD;
    bf16* wob = wvb + DD;
    bf16* q   = wob + DD;       // 4M each
    bf16* k   = q   + MD;
    bf16* v   = k   + MD;
    bf16* ao  = v   + MD;       // total 24M bf16 = 48 MB

    // One-time f32 -> bf16 conversions (memory-bound, ~4 us at 23.3 TB/s).
    cvt_f32_bf16_kernel<<<(int)(MD / 8 / 256), 256, 0, stream>>>(x,  xb,  (int)MD);
    cvt_f32_bf16_kernel<<<(int)(DD / 8 / 256), 256, 0, stream>>>(Wq, wqb, (int)DD);
    cvt_f32_bf16_kernel<<<(int)(DD / 8 / 256), 256, 0, stream>>>(Wk, wkb, (int)DD);
    cvt_f32_bf16_kernel<<<(int)(DD / 8 / 256), 256, 0, stream>>>(Wv, wvb, (int)DD);
    cvt_f32_bf16_kernel<<<(int)(DD / 8 / 256), 256, 0, stream>>>(Wo, wob, (int)DD);

    dim3 gg(D_ / 64, M_ / 64);   // (16, 64)
    dim3 gb(128);
    // QKV projections -> bf16 (B,H,L,DK)
    mha_gemm_bias_kernel<<<gg, gb, 0, stream>>>(xb, wqb, bq, q, M_, D_, D_, 1);
    mha_gemm_bias_kernel<<<gg, gb, 0, stream>>>(xb, wkb, bk, k, M_, D_, D_, 1);
    mha_gemm_bias_kernel<<<gg, gb, 0, stream>>>(xb, wvb, bv, v, M_, D_, D_, 1);
    // Causal flash attention -> bf16 (B,L,D)
    mha_flash_attn_kernel<<<dim3(L_ / 64, B_ * H_), gb, 0, stream>>>(q, k, v, ao);
    // Output projection -> f32 d_out
    mha_gemm_bias_kernel<<<gg, gb, 0, stream>>>(ao, wob, bo, out, M_, D_, D_, 0);
}
